// FAENet_4604204941819
// MI455X (gfx1250) — matrix operand order repaired and checked by reading, not compiled, not run on hardware
//
#include <hip/hip_runtime.h>

// ---------------------------------------------------------------------------
// FAENet forward for MI455X (gfx1250, wave32, WMMA).
// Dominant cost: 4x edge GEMM  ecat[E,384] @ geom_w[384,128]  (~157 GFLOP)
// done with v_wmma_f32_16x16x32_f16, fused swish + message + scatter epilogue.
// Edge GEMM uses persistent blocks + async global->LDS weight staging
// (GLOBAL_LOAD_ASYNC_TO_LDS_B128 / s_wait_asynccnt).  B fragments for a
// k-step are batch-loaded into registers so the WMMAs issue back-to-back.
// ---------------------------------------------------------------------------

typedef _Float16 v8h  __attribute__((ext_vector_type(8)));
typedef _Float16 v16h __attribute__((ext_vector_type(16)));
typedef float    v8f  __attribute__((ext_vector_type(8)));

constexpr int   kN   = 10000;
constexpr int   kE   = 400000;
constexpr int   kG   = 64;          // graphs
constexpr int   kH   = 128;         // hidden
constexpr int   kNF  = 128;         // filters
constexpr int   kKE  = 384;         // NF + 2H  (edge GEMM K)
constexpr int   kNG  = 50;          // gaussians
constexpr int   kNI  = 4;           // interactions
constexpr float kEPS = 1e-5f;
constexpr float kCUT = 6.0f;

constexpr int   kEdgeTiles  = kE / 128;   // 3125 tiles of 128 edges
constexpr int   kEdgeBlocks = 640;        // persistent blocks for edge GEMM

__device__ __forceinline__ float swishf(float x) { return x / (1.0f + __expf(-x)); }

__device__ __forceinline__ v16h cat16(v8h a, v8h b) {
  return __builtin_shufflevector(a, b, 0,1,2,3,4,5,6,7,8,9,10,11,12,13,14,15);
}

// A fragment (16x32 f16, row-major source). lanes<16: K = k0..k0+7, k0+16..k0+23
// lanes>=16: +8 within each 16-group.  (ISA 7.12.2 "16-bit A-Matrix 16x32")
__device__ __forceinline__ v16h load_a_frag(const _Float16* rowPtr, int k0, bool hiHalf) {
  const _Float16* p = rowPtr + k0 + (hiHalf ? 8 : 0);
  return cat16(*(const v8h*)(p), *(const v8h*)(p + 16));
}

// B fragment (32x16 f16) from transposed weights Wt[dout][din]: lane n holds
// column n, K = k0..k0+7, k0+16..k0+23 (lanes>=16: +8) — same order as A.
__device__ __forceinline__ v16h load_b_frag(const _Float16* wtRow, int k0, bool hiHalf) {
  const _Float16* p = wtRow + k0 + (hiHalf ? 8 : 0);
  return cat16(*(const v8h*)(p), *(const v8h*)(p + 16));
}

__device__ __forceinline__ v8f wmma16(v16h a, v16h b, v8f c) {
  return __builtin_amdgcn_wmma_f32_16x16x32_f16(false, a, false, b, (short)0, c, false, false);
}

// Async global->LDS copy of 16 bytes (GLOBAL_LOAD_ASYNC_TO_LDS_B128, ASYNCcnt).
__device__ __forceinline__ void async_lds_b128(unsigned ldsOff, const void* gptr) {
  unsigned long long ga = (unsigned long long)(uintptr_t)gptr;
  asm volatile("global_load_async_to_lds_b128 %0, %1, off"
               :: "v"(ldsOff), "v"(ga) : "memory");
}
__device__ __forceinline__ void wait_asynccnt0() {
  asm volatile("s_wait_asynccnt 0x0" ::: "memory");
}

// ---------------------------------------------------------------------------
// Utility kernels
// ---------------------------------------------------------------------------
__global__ void zero_f32_kernel(float* p, int n) {
  int i = blockIdx.x * 256 + threadIdx.x;
  if (i < n) p[i] = 0.0f;
}

// Wt[m][k] = (f16) W[k][m]   (W is [K][M] row-major, as in the reference)
__global__ void transpose_f16_kernel(const float* __restrict__ W, _Float16* __restrict__ Wt,
                                     int K, int M) {
  int idx = blockIdx.x * 256 + threadIdx.x;
  if (idx >= K * M) return;
  int k = idx / M, m = idx - k * M;
  Wt[(size_t)m * K + k] = (_Float16)W[idx];
}

__global__ void count_kernel(const int* __restrict__ batch, float* __restrict__ counts, int n) {
  int i = blockIdx.x * 256 + threadIdx.x;
  if (i < n) atomicAdd(&counts[batch[i]], 1.0f);
}

// ---------------------------------------------------------------------------
// Edge features: rel_pos, gaussian smearing, tiny 3->64 and 50->64 matvecs
// e[E,128] (f16) = swish([rel_pos@W1+b1 | gauss@W2+b2])
// ---------------------------------------------------------------------------
__global__ void __launch_bounds__(256)
edge_feat_kernel(const float* __restrict__ pos, const int* __restrict__ eidx,
                 const float* __restrict__ w1, const float* __restrict__ b1,
                 const float* __restrict__ w2, const float* __restrict__ b2,
                 _Float16* __restrict__ eF16) {
  __shared__ float sw1[3 * 64];
  __shared__ float sw2[kNG * 64];
  __shared__ float sb1[64];
  __shared__ float sb2[64];
  for (int i = threadIdx.x; i < 3 * 64; i += 256) sw1[i] = w1[i];
  for (int i = threadIdx.x; i < kNG * 64; i += 256) sw2[i] = w2[i];
  for (int i = threadIdx.x; i < 64; i += 256) { sb1[i] = b1[i]; sb2[i] = b2[i]; }
  __syncthreads();

  int e = blockIdx.x * 256 + threadIdx.x;
  if (e >= kE) return;
  int s = eidx[e], d = eidx[kE + e];
  float rx = pos[s * 3 + 0] - pos[d * 3 + 0];
  float ry = pos[s * 3 + 1] - pos[d * 3 + 1];
  float rz = pos[s * 3 + 2] - pos[d * 3 + 2];
  float dist = sqrtf(rx * rx + ry * ry + rz * rz);

  const float delta = kCUT / (float)(kNG - 1);
  const float coeff = -0.5f / (delta * delta);
  float g[kNG];
#pragma unroll
  for (int k = 0; k < kNG; ++k) { float t = dist - (float)k * delta; g[k] = __expf(coeff * t * t); }

  _Float16* erow = eF16 + (size_t)e * kNF;
  for (int j = 0; j < 64; ++j) {
    float v1 = sb1[j] + rx * sw1[j] + ry * sw1[64 + j] + rz * sw1[128 + j];
    float v2 = sb2[j];
#pragma unroll
    for (int k = 0; k < kNG; ++k) v2 += g[k] * sw2[k * 64 + j];
    erow[j]      = (_Float16)swishf(v1);
    erow[64 + j] = (_Float16)swishf(v2);
  }
}

// feat[N,128] f16 = concat(emb[z] (96), tag_emb[tags] (32))
__global__ void node_feat_kernel(const int* __restrict__ z, const int* __restrict__ tags,
                                 const float* __restrict__ emb, const float* __restrict__ tag_emb,
                                 _Float16* __restrict__ feat) {
  int idx = blockIdx.x * 256 + threadIdx.x;
  if (idx >= kN * kH) return;
  int n = idx >> 7, c = idx & 127;
  float v = (c < 96) ? emb[z[n] * 96 + c] : tag_emb[tags[n] * 32 + (c - 96)];
  feat[idx] = (_Float16)v;
}

// ---------------------------------------------------------------------------
// Node GEMM (WMMA): out[N,128] = act(in[N,128] @ W[128,128] + b)
// Wt is the f16-transposed weight [128 out][128 in], staged in 32KB LDS.
// One wave = 16 rows x 128 cols; per k-step: batch-load 8 B fragments,
// then 8 back-to-back WMMAs.  4 k-steps x 8 n-tiles = 32 WMMAs per wave.
// ---------------------------------------------------------------------------
__global__ void __launch_bounds__(256)
node_gemm_kernel(const _Float16* __restrict__ inF16, const _Float16* __restrict__ Wt,
                 const float* __restrict__ bias, float* __restrict__ out32,
                 _Float16* __restrict__ outF16, int nRows) {
  __shared__ _Float16 ldsW[kH * kH];  // 32 KB
  for (int i = threadIdx.x; i < (kH * kH) / 8; i += 256)
    ((v8h*)ldsW)[i] = ((const v8h*)Wt)[i];
  __syncthreads();

  const int lane = threadIdx.x & 31;
  const int wave = threadIdx.x >> 5;
  const bool hiHalf = lane >= 16;
  const int colLane = lane & 15;
  const int rbase = (blockIdx.x * 8 + wave) * 16;

  int myRow = rbase + colLane;
  int ldRow = myRow < nRows ? myRow : (nRows - 1);  // clamp: EXEC stays all-ones
  const _Float16* aRow = inF16 + (size_t)ldRow * kH;

  v8f acc[8] = {};
#pragma unroll
  for (int kk = 0; kk < 4; ++kk) {
    v16h a = load_a_frag(aRow, kk * 32, hiHalf);
    v16h bfr[8];
#pragma unroll
    for (int nt = 0; nt < 8; ++nt)
      bfr[nt] = load_b_frag(ldsW + (size_t)(nt * 16 + colLane) * kH, kk * 32, hiHalf);
#pragma unroll
    for (int nt = 0; nt < 8; ++nt)
      acc[nt] = wmma16(a, bfr[nt], acc[nt]);
  }

#pragma unroll
  for (int nt = 0; nt < 8; ++nt) {
    const int col = nt * 16 + colLane;
    const float bv = bias[col];
#pragma unroll
    for (int v = 0; v < 8; ++v) {
      int r = rbase + v + (hiHalf ? 8 : 0);
      if (r < nRows) {
        float x = swishf(acc[nt][v] + bv);
        out32[(size_t)r * kH + col] = x;
        outF16[(size_t)r * kH + col] = (_Float16)x;
      }
    }
  }
}

// ---------------------------------------------------------------------------
// Edge GEMM (WMMA) + fused message/scatter:
//   m = swish([e | h_src | h_dst] @ geom_w + b);  agg[dst] += h32[src] * m
// WtGeom is f16 [128 out][384 in]; staged in LDS as 2 phases of 64 cols (48KB)
// via async global->LDS b128 copies.  Persistent blocks grid-stride over the
// 3125 edge tiles so each block stages the weights only twice total.
// One wave = 16 edges/tile; per k-step: 4 batched B loads then 4 WMMAs.
// E % 128 == 0, so no row guards needed.
// ---------------------------------------------------------------------------
__global__ void __launch_bounds__(256)
edge_geom_kernel(const _Float16* __restrict__ eF16, const _Float16* __restrict__ hF16,
                 const float* __restrict__ h32, const int* __restrict__ eidx,
                 const _Float16* __restrict__ WtGeom, const float* __restrict__ geomB,
                 float* __restrict__ agg) {
  __shared__ _Float16 ldsW[64 * kKE];  // 48 KB

  const int lane = threadIdx.x & 31;
  const int wave = threadIdx.x >> 5;
  const bool hiHalf = lane >= 16;
  const int colLane = lane & 15;

  for (int phase = 0; phase < 2; ++phase) {
    // --- async-stage 64 output columns of Wt (64 x 384 f16 = 48KB) ---
    {
      const _Float16* gsrc = WtGeom + (size_t)(phase * 64) * kKE;
      for (int i = threadIdx.x; i < (64 * kKE) / 8; i += 256) {
        unsigned ldsOff = (unsigned)(uintptr_t)(const void*)&ldsW[i * 8];
        async_lds_b128(ldsOff, gsrc + (size_t)i * 8);
      }
      wait_asynccnt0();
    }
    __syncthreads();

    // bias for this phase's 64 columns (invariant across tiles)
    float bv[4];
#pragma unroll
    for (int nt = 0; nt < 4; ++nt) bv[nt] = geomB[phase * 64 + nt * 16 + colLane];

    // --- grid-stride over edge tiles (128 edges per tile, 16 per wave) ---
    for (int t = blockIdx.x; t < kEdgeTiles; t += gridDim.x) {
      const int base = (t * 8 + wave) * 16;
      const int myEdge = base + colLane;

      const int srcMy = eidx[myEdge];
      const int dstMy = eidx[kE + myEdge];
      const _Float16* p0 = eF16 + (size_t)myEdge * kNF;
      const _Float16* p1 = hF16 + (size_t)srcMy * kH;
      const _Float16* p2 = hF16 + (size_t)dstMy * kH;

      v8f acc[4] = {};
#pragma unroll
      for (int kk = 0; kk < 12; ++kk) {
        const int seg = kk >> 2;            // 0: e, 1: h[src], 2: h[dst]
        const int kin = (kk & 3) * 32;      // offset within the 128-wide segment
        const _Float16* sp = (seg == 0) ? p0 : ((seg == 1) ? p1 : p2);
        v16h a = load_a_frag(sp, kin, hiHalf);
        v16h bfr[4];
#pragma unroll
        for (int nt = 0; nt < 4; ++nt)
          bfr[nt] = load_b_frag(ldsW + (size_t)(nt * 16 + colLane) * kKE, kk * 32, hiHalf);
#pragma unroll
        for (int nt = 0; nt < 4; ++nt)
          acc[nt] = wmma16(a, bfr[nt], acc[nt]);
      }

      // fused epilogue: bias + swish, message = h32[src]*m, scatter atomicAdd
#pragma unroll
      for (int v = 0; v < 8; ++v) {
        const int rloc = v + (hiHalf ? 8 : 0);            // local edge row 0..15
        const int s = __shfl(srcMy, rloc, 32);
        const int d = __shfl(dstMy, rloc, 32);
        const float* hs = h32 + (size_t)s * kH;
        float* ag = agg + (size_t)d * kH;
#pragma unroll
        for (int nt = 0; nt < 4; ++nt) {
          const int col = phase * 64 + nt * 16 + colLane;
          const float m = swishf(acc[nt][v] + bv[nt]);
          atomicAdd(&ag[col], hs[col] * m);
        }
      }
    }
    __syncthreads();  // all waves done with ldsW before phase-1 restage
  }
}

// ---------------------------------------------------------------------------
// GraphNorm (3 passes) — per-graph segment stats via atomics
// ---------------------------------------------------------------------------
__global__ void gn_sum_kernel(const float* __restrict__ agg, const int* __restrict__ batch,
                              float* __restrict__ gnSum) {
  int idx = blockIdx.x * 256 + threadIdx.x;
  if (idx >= kN * kH) return;
  int node = idx >> 7, c = idx & 127;
  atomicAdd(&gnSum[batch[node] * kH + c], agg[idx]);
}

__global__ void gn_center_kernel(const float* __restrict__ agg, const int* __restrict__ batch,
                                 const float* __restrict__ counts, const float* __restrict__ gnSum,
                                 const float* __restrict__ meanScale, float* __restrict__ x1,
                                 float* __restrict__ gnVar) {
  int idx = blockIdx.x * 256 + threadIdx.x;
  if (idx >= kN * kH) return;
  int node = idx >> 7, c = idx & 127;
  int g = batch[node];
  float cnt = fmaxf(counts[g], 1.0f);
  float mean = gnSum[g * kH + c] / cnt;
  float x = agg[idx] - mean * meanScale[c];
  x1[idx] = x;
  atomicAdd(&gnVar[g * kH + c], x * x);
}

__global__ void gn_final_kernel(const float* __restrict__ x1, const int* __restrict__ batch,
                                const float* __restrict__ counts, const float* __restrict__ gnVar,
                                const float* __restrict__ w, const float* __restrict__ b,
                                _Float16* __restrict__ outF16) {
  int idx = blockIdx.x * 256 + threadIdx.x;
  if (idx >= kN * kH) return;
  int node = idx >> 7, c = idx & 127;
  int g = batch[node];
  float cnt = fmaxf(counts[g], 1.0f);
  float var = gnVar[g * kH + c] / cnt;
  float y = x1[idx] * rsqrtf(var + kEPS) * w[c] + b[c];
  outF16[idx] = (_Float16)swishf(y);
}

// ---------------------------------------------------------------------------
// Output block: skips[g][i] += (swish(h@l1)+b)@l2 * (h@wlin+b)  per node
// ---------------------------------------------------------------------------
__global__ void __launch_bounds__(256)
output_block_kernel(const float* __restrict__ h32, const int* __restrict__ batch,
                    const float* __restrict__ wlin, const float* __restrict__ wlinB,
                    const float* __restrict__ l1w, const float* __restrict__ l1b,
                    const float* __restrict__ l2w, const float* __restrict__ l2b,
                    float* __restrict__ skips, int skipIdx) {
  __shared__ float s1[kH * 64];
  __shared__ float swl[kH];
  __shared__ float s2[64];
  __shared__ float sl1b[64];
  for (int i = threadIdx.x; i < kH * 64; i += 256) s1[i] = l1w[i];
  for (int i = threadIdx.x; i < kH; i += 256) swl[i] = wlin[i];
  for (int i = threadIdx.x; i < 64; i += 256) { s2[i] = l2w[i]; sl1b[i] = l1b[i]; }
  __syncthreads();

  int n = blockIdx.x * 256 + threadIdx.x;
  if (n >= kN) return;
  const float* hp = h32 + (size_t)n * kH;
  float hr[kH];
#pragma unroll
  for (int c = 0; c < kH; ++c) hr[c] = hp[c];

  float alpha = wlinB[0];
#pragma unroll
  for (int c = 0; c < kH; ++c) alpha += hr[c] * swl[c];

  float hh2 = l2b[0];
  for (int j = 0; j < 64; ++j) {
    float a = sl1b[j];
#pragma unroll
    for (int c = 0; c < kH; ++c) a += hr[c] * s1[c * 64 + j];
    hh2 += swishf(a) * s2[j];
  }
  atomicAdd(&skips[batch[n] * (kNI + 1) + skipIdx], hh2 * alpha);
}

__global__ void energy_kernel(const float* __restrict__ skips, const float* __restrict__ skipW,
                              const float* __restrict__ skipB, float* __restrict__ dout) {
  int g = threadIdx.x;
  if (g < kG) {
    float e = skipB[0];
#pragma unroll
    for (int i = 0; i < kNI + 1; ++i) e += skips[g * (kNI + 1) + i] * skipW[i];
    dout[g] = e;
  }
}

__global__ void copy_h_kernel(const float* __restrict__ h32, float* __restrict__ dout) {
  int idx = blockIdx.x * 256 + threadIdx.x;
  if (idx < kN * kH) dout[idx] = h32[idx];
}

// ---------------------------------------------------------------------------
// Host launch
// ---------------------------------------------------------------------------
extern "C" void kernel_launch(void* const* d_in, const int* in_sizes, int n_in,
                              void* d_out, int out_size, void* d_ws, size_t ws_size,
                              hipStream_t stream) {
  (void)in_sizes; (void)n_in; (void)out_size; (void)ws_size;

  const int*   z       = (const int*)d_in[0];
  const float* pos     = (const float*)d_in[1];
  const int*   tags    = (const int*)d_in[2];
  const int*   batch   = (const int*)d_in[3];
  const int*   eidx    = (const int*)d_in[4];
  const float* emb     = (const float*)d_in[5];
  const float* tag_emb = (const float*)d_in[6];
  const float* lin_w   = (const float*)d_in[7];
  const float* lin_b   = (const float*)d_in[8];
  const float* e1w     = (const float*)d_in[9];
  const float* e1b     = (const float*)d_in[10];
  const float* e2w     = (const float*)d_in[11];
  const float* e2b     = (const float*)d_in[12];
  auto ip = [&](int i, int j) { return (const float*)d_in[13 + i * 9 + j]; };
  const float* out_l1w  = (const float*)d_in[49];
  const float* out_l1b  = (const float*)d_in[50];
  const float* out_l2w  = (const float*)d_in[51];
  const float* out_l2b  = (const float*)d_in[52];
  const float* out_wlw  = (const float*)d_in[53];
  const float* out_wlb  = (const float*)d_in[54];
  const float* skip_w   = (const float*)d_in[55];
  const float* skip_b   = (const float*)d_in[56];

  // workspace bump allocator (256B aligned)
  char* wsp = (char*)d_ws;
  auto alloc = [&](size_t bytes) -> void* {
    void* p = (void*)wsp;
    wsp += (bytes + 255) & ~(size_t)255;
    return p;
  };
  _Float16* eF16    = (_Float16*)alloc((size_t)kE * kNF * 2);  // 102.4 MB (L2-resident)
  _Float16* featF16 = (_Float16*)alloc((size_t)kN * kH * 2);
  _Float16* hF16    = (_Float16*)alloc((size_t)kN * kH * 2);
  _Float16* tAF16   = (_Float16*)alloc((size_t)kN * kH * 2);
  _Float16* tBF16   = (_Float16*)alloc((size_t)kN * kH * 2);
  float*    h32     = (float*)alloc((size_t)kN * kH * 4);
  float*    x1_32   = (float*)alloc((size_t)kN * kH * 4);
  float*    tB32    = (float*)alloc((size_t)kN * kH * 4);
  float*    agg     = (float*)alloc((size_t)kN * kH * 4);
  float*    gnSum   = (float*)alloc((size_t)kG * kH * 4);
  float*    gnVar   = (float*)alloc((size_t)kG * kH * 4);
  float*    counts  = (float*)alloc((size_t)kG * 4);
  float*    skips   = (float*)alloc((size_t)kG * (kNI + 1) * 4);
  _Float16* WtLin   = (_Float16*)alloc((size_t)kH * kH * 2);
  _Float16* WtGeom  = (_Float16*)alloc((size_t)kNI * kH * kKE * 2);
  _Float16* WtLinH  = (_Float16*)alloc((size_t)kNI * kH * kH * 2);
  _Float16* WtOther = (_Float16*)alloc((size_t)kNI * kH * kH * 2);

  auto cdiv = [](int a, int b) { return (a + b - 1) / b; };
  const int NH = kN * kH;

  // --- prep: zero atomic targets, counts, weight transposes ---
  zero_f32_kernel<<<cdiv(kG * (kNI + 1), 256), 256, 0, stream>>>(skips, kG * (kNI + 1));
  zero_f32_kernel<<<1, 256, 0, stream>>>(counts, kG);
  count_kernel<<<cdiv(kN, 256), 256, 0, stream>>>(batch, counts, kN);

  transpose_f16_kernel<<<cdiv(kH * kH, 256), 256, 0, stream>>>(lin_w, WtLin, kH, kH);
  for (int i = 0; i < kNI; ++i) {
    transpose_f16_kernel<<<cdiv(kKE * kH, 256), 256, 0, stream>>>(
        ip(i, 0), WtGeom + (size_t)i * kH * kKE, kKE, kH);
    transpose_f16_kernel<<<cdiv(kH * kH, 256), 256, 0, stream>>>(
        ip(i, 2), WtLinH + (size_t)i * kH * kH, kH, kH);
    transpose_f16_kernel<<<cdiv(kH * kH, 256), 256, 0, stream>>>(
        ip(i, 4), WtOther + (size_t)i * kH * kH, kH, kH);
  }

  // --- edge features + node embedding ---
  edge_feat_kernel<<<cdiv(kE, 256), 256, 0, stream>>>(pos, eidx, e1w, e1b, e2w, e2b, eF16);
  node_feat_kernel<<<cdiv(NH, 256), 256, 0, stream>>>(z, tags, emb, tag_emb, featF16);
  node_gemm_kernel<<<cdiv(kN, 128), 256, 0, stream>>>(featF16, WtLin, lin_b, h32, hF16, kN);

  // --- interactions ---
  for (int i = 0; i < kNI; ++i) {
    output_block_kernel<<<cdiv(kN, 256), 256, 0, stream>>>(
        h32, batch, out_wlw, out_wlb, out_l1w, out_l1b, out_l2w, out_l2b, skips, i);

    zero_f32_kernel<<<cdiv(NH, 256), 256, 0, stream>>>(agg, NH);
    edge_geom_kernel<<<kEdgeBlocks, 256, 0, stream>>>(
        eF16, hF16, h32, eidx, WtGeom + (size_t)i * kH * kKE, ip(i, 1), agg);

    zero_f32_kernel<<<cdiv(kG * kH, 256), 256, 0, stream>>>(gnSum, kG * kH);
    zero_f32_kernel<<<cdiv(kG * kH, 256), 256, 0, stream>>>(gnVar, kG * kH);
    gn_sum_kernel<<<cdiv(NH, 256), 256, 0, stream>>>(agg, batch, gnSum);
    gn_center_kernel<<<cdiv(NH, 256), 256, 0, stream>>>(
        agg, batch, counts, gnSum, ip(i, 8), x1_32, gnVar);
    gn_final_kernel<<<cdiv(NH, 256), 256, 0, stream>>>(
        x1_32, batch, counts, gnVar, ip(i, 6), ip(i, 7), tAF16);

    node_gemm_kernel<<<cdiv(kN, 128), 256, 0, stream>>>(
        tAF16, WtLinH + (size_t)i * kH * kH, ip(i, 3), tB32, tBF16, kN);
    node_gemm_kernel<<<cdiv(kN, 128), 256, 0, stream>>>(
        tBF16, WtOther + (size_t)i * kH * kH, ip(i, 5), h32, hF16, kN);
  }

  // --- final output block, energy, h copy ---
  output_block_kernel<<<cdiv(kN, 256), 256, 0, stream>>>(
      h32, batch, out_wlw, out_wlb, out_l1w, out_l1b, out_l2w, out_l2b, skips, kNI);

  float* dout = (float*)d_out;
  energy_kernel<<<1, 64, 0, stream>>>(skips, skip_w, skip_b, dout);
  copy_h_kernel<<<cdiv(NH, 256), 256, 0, stream>>>(h32, dout + kG);
}
